// MambaEncoder_88897233092708
// MI455X (gfx1250) — matrix operand order repaired and checked by reading, not compiled, not run on hardware
//
#include <hip/hip_runtime.h>
#include <hip/hip_bf16.h>

typedef __attribute__((ext_vector_type(16))) _Float16 v16h;
typedef __attribute__((ext_vector_type(8)))  float    v8f;

#define BB        8
#define LL        2048
#define EE        512
#define HH        512
#define DSTATE    64
#define HEADDIM   64
#define DINNER    1024
#define NHEADS    16
#define CONVDIM   1152
#define DPROJ     2192
#define DCONV     4
#define EPSF      1e-5f
#define BL        (BB*LL)

__device__ __forceinline__ float silu_f(float x) { return x / (1.0f + __expf(-x)); }

// ---------------------------------------------------------------------------
// Generic WMMA GEMM: C[m, coff + n] = sum_k A[rowA(m), k] * W[n, k]  (+bias[n])
// A: row-major MxK (fp32, gathered rows via rowA), W: row-major NxK (fp32).
// f32 -> f16 in registers, f32 accumulation via v_wmma_f32_16x16x32_f16.
// Block = 128 threads = 4 waves; each wave owns a 16(M) x 64(N) strip
// (4 accumulators, A-fragment reused across the 4 N-subtiles).
//
// Ragged N (only N=2192 here) is handled by CLAMPING the subtile column base
// to N-16: out-of-range subtiles redundantly recompute the last valid tile
// and store identical values.  This keeps the K-loop 100% branch-free so the
// accumulators stay pinned in fixed VGPRs (no v_mov shuffling, no per-iter
// scalar branches).  M is always a multiple of 64 (M = B*L); N a multiple of 16.
// ---------------------------------------------------------------------------
__global__ __launch_bounds__(128)
void wmma_gemm_f16(const float* __restrict__ A, const int* __restrict__ rowA,
                   const float* __restrict__ W, const float* __restrict__ bias,
                   float* __restrict__ C, const int* __restrict__ rowC,
                   int N, int K, int ldc, int coff)
{
    const int lane  = threadIdx.x & 31;
    const int wave  = threadIdx.x >> 5;
    const int Mbase = blockIdx.y * 64 + wave * 16;
    const int Nbase = blockIdx.x * 64;
    const int hl    = lane & 15;        // position within half-wave
    const int hi    = lane >> 4;        // upper half-wave?

    // Clamped subtile column bases (branch-free ragged-edge handling).
    int ncol[4];
    #pragma unroll
    for (int t = 0; t < 4; ++t) {
        int c = Nbase + 16 * t;
        ncol[t] = (c <= N - 16) ? c : (N - 16);
    }

    // A-fragment source row (contiguous K order matching B: lanes 0-15 supply
    // M=lane with K 0..15; lanes 16-31 supply M=lane-16 with K 16..31).
    const int  am   = Mbase + hl;
    const long arow = rowA ? (long)rowA[am] : (long)am;
    const float* aptr = A + arow * (long)K + hi * 16;

    // B-fragment base pointers (32x16 f16: lane -> column N, K-run of 16).
    const float* wptr[4];
    #pragma unroll
    for (int t = 0; t < 4; ++t)
        wptr[t] = W + (long)(ncol[t] + hl) * K + hi * 16;

    v8f acc[4];
    #pragma unroll
    for (int t = 0; t < 4; ++t)
        #pragma unroll
        for (int r = 0; r < 8; ++r) acc[t][r] = 0.0f;

    for (int kk = 0; kk < K; kk += 32) {
        __builtin_prefetch(aptr + kk + 64, 0, 1);   // speculative global_prefetch_b8

        // ---- A fragment (16x32 f16) ------------------------------------
        __align__(16) float atmp[16];
        {
            const float4* p0 = reinterpret_cast<const float4*>(aptr + kk);
            const float4* p1 = reinterpret_cast<const float4*>(aptr + kk + 8);
            ((float4*)atmp)[0] = p0[0];
            ((float4*)atmp)[1] = p0[1];
            ((float4*)atmp)[2] = p1[0];
            ((float4*)atmp)[3] = p1[1];
        }
        v16h afrag;
        #pragma unroll
        for (int i = 0; i < 16; ++i) afrag[i] = (_Float16)atmp[i];

        // ---- 4 B fragments + 4 back-to-back WMMAs ----------------------
        #pragma unroll
        for (int t = 0; t < 4; ++t) {
            const float4* wp = reinterpret_cast<const float4*>(wptr[t] + kk);
            __align__(16) float btmp[16];
            ((float4*)btmp)[0] = wp[0];
            ((float4*)btmp)[1] = wp[1];
            ((float4*)btmp)[2] = wp[2];
            ((float4*)btmp)[3] = wp[3];
            v16h bfrag;
            #pragma unroll
            for (int i = 0; i < 16; ++i) bfrag[i] = (_Float16)btmp[i];

            acc[t] = __builtin_amdgcn_wmma_f32_16x16x32_f16(
                         false, afrag, false, bfrag,
                         (short)0, acc[t], false, false);
        }
    }

    // ---- Epilogue: f32 C/D layout — VGPR r holds (M = r + 8*hi, N = hl) ----
    #pragma unroll
    for (int t = 0; t < 4; ++t) {
        const int col = ncol[t] + hl;
        const float bv = bias ? bias[col] : 0.0f;
        #pragma unroll
        for (int r = 0; r < 8; ++r) {
            const int  row  = Mbase + r + hi * 8;
            const long crow = rowC ? (long)rowC[row] : (long)row;
            C[crow * (long)ldc + coff + col] = acc[t][r] + bv;
        }
    }
}

// ---------------------------------------------------------------------------
// Length-flip map (involution): fmap[b*L + t] = b*L + (t < len ? len-1-t : t)
// ---------------------------------------------------------------------------
__global__ void flipmap_kernel(const int* __restrict__ lens, int* __restrict__ fmap)
{
    int m = blockIdx.x * blockDim.x + threadIdx.x;
    if (m >= BL) return;
    int b = m / LL, t = m % LL;
    int len = lens[b];
    int idx = (t < len) ? (len - 1 - t) : t;
    fmap[m] = b * LL + idx;
}

// ---------------------------------------------------------------------------
// Depthwise causal conv (D_CONV=4 taps) + bias + SiLU over the xBC slice of
// zxbcdt.  One thread per (b, l, c); channel-fastest for coalescing.
// ---------------------------------------------------------------------------
__global__ void conv_silu_kernel(const float* __restrict__ zx,
                                 const float* __restrict__ cw,
                                 const float* __restrict__ cb,
                                 float* __restrict__ xbc)
{
    long idx = (long)blockIdx.x * blockDim.x + threadIdx.x;
    if (idx >= (long)BL * CONVDIM) return;
    int  c  = (int)(idx % CONVDIM);
    long bl = idx / CONVDIM;
    int  l  = (int)(bl % LL);
    long base = bl * (long)DPROJ + DINNER + c;   // xBC slice starts at DINNER
    float acc = cb[c];
    #pragma unroll
    for (int j = 0; j < DCONV; ++j) {
        int ls = l - (DCONV - 1) + j;
        if (ls >= 0) acc += cw[c * DCONV + j] * zx[base + (long)(ls - l) * DPROJ];
    }
    xbc[bl * (long)CONVDIM + c] = silu_f(acc);
}

// ---------------------------------------------------------------------------
// Mamba2 recurrence.  One workgroup per (batch, head); 64x64 state fully
// register-resident: thread tid owns row p = tid>>2, state columns
// [16*(tid&3), 16*(tid&3)+16).  B_t, C_t, x_t staged in LDS per step; the
// C-contraction is reduced across the 4 lanes sharing p via shfl_xor.
// ---------------------------------------------------------------------------
__global__ __launch_bounds__(256)
void mamba_scan_kernel(const float* __restrict__ zx, const float* __restrict__ xbc,
                       const float* __restrict__ dtb, const float* __restrict__ alog,
                       const float* __restrict__ Dp, float* __restrict__ y)
{
    const int wg   = blockIdx.x;
    const int b    = wg / NHEADS;
    const int head = wg % NHEADS;
    const int tid  = threadIdx.x;
    const int p    = tid >> 2;
    const int nq   = tid & 3;

    __shared__ float sB[DSTATE], sC[DSTATE], sX[HEADDIM];

    float s[16];
    #pragma unroll
    for (int i = 0; i < 16; ++i) s[i] = 0.0f;

    const float dtbias = dtb[head];
    const float Aexp   = __expf(alog[head]);
    const float Dv     = Dp[head];

    for (int t = 0; t < LL; ++t) {
        const long row = (long)b * LL + t;
        const long xb  = row * (long)CONVDIM;
        __syncthreads();
        if (tid < 64)        sB[tid]       = xbc[xb + DINNER + tid];
        else if (tid < 128)  sC[tid - 64]  = xbc[xb + DINNER + DSTATE + (tid - 64)];
        else if (tid < 192)  sX[tid - 128] = xbc[xb + head * HEADDIM + (tid - 128)];
        __syncthreads();

        float dtraw = zx[row * (long)DPROJ + (DINNER + CONVDIM) + head] + dtbias;
        float dt    = (dtraw > 20.0f) ? dtraw : log1pf(__expf(dtraw));   // softplus
        float dA    = __expf(-Aexp * dt);
        float xp    = sX[p];
        float coef  = dt * xp;

        float accv = 0.0f;
        #pragma unroll
        for (int i = 0; i < 16; ++i) {
            float sv = s[i] * dA + coef * sB[nq * 16 + i];
            s[i] = sv;
            accv += sv * sC[nq * 16 + i];
        }
        accv += __shfl_xor(accv, 1, 32);
        accv += __shfl_xor(accv, 2, 32);
        if (nq == 0)
            y[row * (long)DINNER + head * HEADDIM + p] = accv + Dv * xp;
    }
}

// ---------------------------------------------------------------------------
// Gated RMSNorm: y = (y * silu(z)) * rsqrt(mean(g^2)+eps) * norm_w, in place.
// One workgroup per (b,l) row; 4 elements/thread over D_INNER=1024.
// ---------------------------------------------------------------------------
__global__ __launch_bounds__(256)
void gated_norm_kernel(float* __restrict__ y, const float* __restrict__ zx,
                       const float* __restrict__ nw)
{
    const long row = blockIdx.x;
    const int  tid = threadIdx.x;
    __shared__ float red[256];

    float g[4];
    float ss = 0.0f;
    #pragma unroll
    for (int i = 0; i < 4; ++i) {
        int   c  = tid + i * 256;
        float yv = y[row * DINNER + c];
        float zv = zx[row * (long)DPROJ + c];      // z occupies cols [0, D_INNER)
        float gv = yv * silu_f(zv);
        g[i] = gv;
        ss += gv * gv;
    }
    red[tid] = ss;
    __syncthreads();
    for (int off = 128; off > 0; off >>= 1) {
        if (tid < off) red[tid] += red[tid + off];
        __syncthreads();
    }
    const float scale = rsqrtf(red[0] / (float)DINNER + EPSF);
    #pragma unroll
    for (int i = 0; i < 4; ++i) {
        int c = tid + i * 256;
        y[row * DINNER + c] = g[i] * scale * nw[c];
    }
}

// ---------------------------------------------------------------------------
// Residual add + LayerNorm over H=512, in place on h.
// ---------------------------------------------------------------------------
__global__ __launch_bounds__(256)
void add_ln_kernel(float* __restrict__ h, const float* __restrict__ tmp,
                   const float* __restrict__ g, const float* __restrict__ bta)
{
    const long row = blockIdx.x;
    const int  tid = threadIdx.x;
    __shared__ float r1[256], r2[256];

    float v[2];
    float sm = 0.0f, sq = 0.0f;
    #pragma unroll
    for (int i = 0; i < 2; ++i) {
        int   c  = tid + i * 256;
        float vv = h[row * HH + c] + tmp[row * HH + c];
        v[i] = vv;
        sm += vv;
        sq += vv * vv;
    }
    r1[tid] = sm; r2[tid] = sq;
    __syncthreads();
    for (int off = 128; off > 0; off >>= 1) {
        if (tid < off) { r1[tid] += r1[tid + off]; r2[tid] += r2[tid + off]; }
        __syncthreads();
    }
    const float mu  = r1[0] / (float)HH;
    const float var = r2[0] / (float)HH - mu * mu;
    const float inv = rsqrtf(var + EPSF);
    #pragma unroll
    for (int i = 0; i < 2; ++i) {
        int c = tid + i * 256;
        h[row * HH + c] = (v[i] - mu) * inv * g[c] + bta[c];
    }
}

// ---------------------------------------------------------------------------
// Masked mean pool over valid lengths: pooled[b, :] (2H = 1024 cols).
// ---------------------------------------------------------------------------
__global__ __launch_bounds__(256)
void pool_kernel(const float* __restrict__ enc, const int* __restrict__ lens,
                 float* __restrict__ pooled)
{
    const int b   = blockIdx.x;
    const int tid = threadIdx.x;
    const int len = lens[b];
    const float denom = (len > 0) ? (float)len : 1.0f;
    float acc[4] = {0.0f, 0.0f, 0.0f, 0.0f};
    for (int l = 0; l < len; ++l) {
        const long base = ((long)b * LL + l) * (2 * HH);
        #pragma unroll
        for (int i = 0; i < 4; ++i) acc[i] += enc[base + tid + i * 256];
    }
    #pragma unroll
    for (int i = 0; i < 4; ++i)
        pooled[(long)b * (2 * HH) + tid + i * 256] = acc[i] / denom;
}

// ---------------------------------------------------------------------------
// Adapter head: hidden[b, o] = tanh(b_ad[o] + pooled[b, :] . W_ad[o, :])
// ---------------------------------------------------------------------------
__global__ void head_kernel(const float* __restrict__ pooled,
                            const float* __restrict__ Wad,
                            const float* __restrict__ bad,
                            float* __restrict__ out)
{
    int n = blockIdx.x * blockDim.x + threadIdx.x;
    if (n >= BB * HH) return;
    int b = n / HH, o = n % HH;
    float acc = bad[o];
    const float* w  = Wad + (long)o * (2 * HH);
    const float* pv = pooled + (long)b * (2 * HH);
    for (int k = 0; k < 2 * HH; ++k) acc += pv[k] * w[k];
    out[n] = tanhf(acc);
}

// ---------------------------------------------------------------------------
// Host-side orchestration
// ---------------------------------------------------------------------------
static inline void launch_gemm(const float* A, const int* rA, const float* W,
                               const float* bias, float* C, const int* rC,
                               int N, int K, int ldc, int coff, hipStream_t s)
{
    dim3 grid((N + 63) / 64, BL / 64);
    wmma_gemm_f16<<<grid, 128, 0, s>>>(A, rA, W, bias, C, rC, N, K, ldc, coff);
}

extern "C" void kernel_launch(void* const* d_in, const int* in_sizes, int n_in,
                              void* d_out, int out_size, void* d_ws, size_t ws_size,
                              hipStream_t stream)
{
    (void)in_sizes; (void)n_in; (void)out_size; (void)ws_size;

    const int*   tokens = (const int*)  d_in[0];
    const int*   lens   = (const int*)  d_in[1];
    const float* emb    = (const float*)d_in[2];
    const float* W_inp  = (const float*)d_in[3];
    const float* b_inp  = (const float*)d_in[4];
    const float* m_Win  = (const float*)d_in[5];
    const float* m_cw   = (const float*)d_in[6];
    const float* m_cb   = (const float*)d_in[7];
    const float* m_dtb  = (const float*)d_in[8];
    const float* m_Al   = (const float*)d_in[9];
    const float* m_D    = (const float*)d_in[10];
    const float* m_nw   = (const float*)d_in[11];
    const float* m_Wo   = (const float*)d_in[12];
    const float* bWo    = (const float*)d_in[13];
    const float* bbo    = (const float*)d_in[14];
    const float* lng    = (const float*)d_in[15];
    const float* lnb    = (const float*)d_in[16];
    const float* Wenc   = (const float*)d_in[17];
    const float* benc   = (const float*)d_in[18];
    const float* Wad    = (const float*)d_in[19];
    const float* bad    = (const float*)d_in[20];

    float* out     = (float*)d_out;
    float* enc_out = out;                           // (B, L, 2H)
    float* hid_out = out + (size_t)BL * (2 * HH);   // (1, B, H)

    // Workspace layout (fp32 elements)
    float* ws     = (float*)d_ws;
    float* h      = ws;                               // BL *  512
    float* zx     = h      + (size_t)BL * HH;         // BL * 2192
    float* xbc    = zx     + (size_t)BL * DPROJ;      // BL * 1152
    float* ybuf   = xbc    + (size_t)BL * CONVDIM;    // BL * 1024
    float* comb   = ybuf   + (size_t)BL * DINNER;     // BL * 1024
    float* tmp    = comb   + (size_t)BL * DINNER;     // BL *  512
    float* pooled = tmp    + (size_t)BL * HH;         // B  * 1024
    int*   fmap   = (int*)(pooled + (size_t)BB * (2 * HH));

    // 0) flip map (length-involution)
    flipmap_kernel<<<(BL + 255) / 256, 256, 0, stream>>>(lens, fmap);

    // 1) embedding gather + input projection:  h = emb[tokens] @ W_inp^T + b_inp
    launch_gemm(emb, tokens, W_inp, b_inp, h, nullptr, HH, EE, HH, 0, stream);

    // 2) layers
    for (int i = 0; i < 4; ++i) {
        for (int d = 0; d < 2; ++d) {
            const int id = i * 2 + d;
            const float* Win = m_Win + (size_t)id * DPROJ * HH;
            const float* cw  = m_cw  + (size_t)id * CONVDIM * DCONV;
            const float* cb  = m_cb  + (size_t)id * CONVDIM;
            const float* dtb = m_dtb + (size_t)id * NHEADS;
            const float* Al  = m_Al  + (size_t)id * NHEADS;
            const float* Dpp = m_D   + (size_t)id * NHEADS;
            const float* nw  = m_nw  + (size_t)id * DINNER;
            const float* Wo  = m_Wo  + (size_t)id * HH * DINNER;

            // in-projection (bwd direction gathers flipped rows)
            launch_gemm(h, d ? fmap : nullptr, Win, nullptr,
                        zx, nullptr, DPROJ, HH, DPROJ, 0, stream);

            // depthwise causal conv + SiLU on xBC slice
            {
                long nthr = (long)BL * CONVDIM;
                conv_silu_kernel<<<(unsigned)((nthr + 255) / 256), 256, 0, stream>>>(
                    zx, cw, cb, xbc);
            }

            // SSM recurrence
            mamba_scan_kernel<<<BB * NHEADS, 256, 0, stream>>>(
                zx, xbc, dtb, Al, Dpp, ybuf);

            // gated RMSNorm (in place on ybuf)
            gated_norm_kernel<<<BL, 256, 0, stream>>>(ybuf, zx, nw);

            // out-projection; bwd scatters flipped rows into comb[:, 512:]
            launch_gemm(ybuf, nullptr, Wo, nullptr,
                        comb, d ? fmap : nullptr, HH, DINNER, 2 * HH,
                        d ? HH : 0, stream);
        }

        // block mixer + residual LayerNorm
        launch_gemm(comb, nullptr, bWo + (size_t)i * HH * (2 * HH),
                    bbo + (size_t)i * HH, tmp, nullptr, HH, 2 * HH, HH, 0, stream);
        add_ln_kernel<<<BL, 256, 0, stream>>>(h, tmp,
                                              lng + (size_t)i * HH,
                                              lnb + (size_t)i * HH);
    }

    // 3) encoder output projection directly into d_out
    launch_gemm(h, nullptr, Wenc, benc, enc_out, nullptr, 2 * HH, HH, 2 * HH, 0, stream);

    // 4) masked mean pool + tanh adapter head
    pool_kernel<<<BB, 256, 0, stream>>>(enc_out, lens, pooled);
    head_kernel<<<(BB * HH + 255) / 256, 256, 0, stream>>>(pooled, Wad, bad, hid_out);
}
